// CandidateAwareClock_36103495090263
// MI455X (gfx1250) — matrix operand order repaired and checked by reading, not compiled, not run on hardware
//
#include <hip/hip_runtime.h>
#include <hip/hip_bf16.h>

// ---------------------------------------------------------------------------
// CandidateAwareClock, algebraically reduced:
//   logits[b,l] = (q[b]@Wk) . token[b,l] / tau     (bk term cancels in softmax)
//   u[b]        = Wv @ (sum_l attn[b,l]*token[b,l]) + bv
// GEMMs: v_wmma_f32_16x16x32_bf16 (wave32, 16x16 tiles).
// Token tile staging: TDM tensor_load_to_lds (200KB/block, 1-D descriptor).
// ---------------------------------------------------------------------------

#define BDIM 256
#define LDIM 200
#define DDIM 512
#define NUMDT 168

typedef __attribute__((ext_vector_type(16))) __bf16 v16bf;
typedef __attribute__((ext_vector_type(8)))  float  v8f;
typedef __attribute__((ext_vector_type(4)))  unsigned v4u;
typedef __attribute__((ext_vector_type(8)))  int      v8i;
typedef __attribute__((ext_vector_type(4)))  int      v4i;

__device__ __forceinline__ __bf16 f2bf(float f) {
  unsigned u = __builtin_bit_cast(unsigned, f);
  unsigned r = u + 0x7FFFu + ((u >> 16) & 1u);       // round-to-nearest-even
  unsigned short h = (unsigned short)(r >> 16);
  return __builtin_bit_cast(__bf16, h);
}
__device__ __forceinline__ float bf2f(__bf16 h) {
  unsigned u = ((unsigned)__builtin_bit_cast(unsigned short, h)) << 16;
  return __builtin_bit_cast(float, u);
}

// --------------------------- prep: weights / q -----------------------------
__global__ void prep_weights(const float* __restrict__ Wk,
                             const float* __restrict__ Wv,
                             const float* __restrict__ item_table,
                             const int*   __restrict__ cand,
                             __bf16* __restrict__ WkT_bf,
                             __bf16* __restrict__ Wv_bf,
                             __bf16* __restrict__ q_bf) {
  int t = blockIdx.x * blockDim.x + threadIdx.x;       // 0 .. 262143
  int n = t >> 9, i = t & 511;
  WkT_bf[t] = f2bf(Wk[i * DDIM + n]);
  Wv_bf[t]  = f2bf(Wv[t]);
  if (t < BDIM * DDIM) {
    int b = t >> 9, d = t & 511;
    q_bf[t] = f2bf(item_table[(size_t)cand[b] * DDIM + d]);
  }
}

// ------------------------------ token build --------------------------------
__global__ void token_build(const int* __restrict__ items,
                            const int* __restrict__ dts,
                            const float* __restrict__ item_table,
                            const float* __restrict__ dt_table,
                            const float* __restrict__ Wc,
                            const float* __restrict__ bc,
                            __bf16* __restrict__ token_bf) {
  int row  = blockIdx.x;                               // 0 .. B*L-1
  int item = items[row];
  int dt   = dts[row];
  float ang = 6.283185307179586f * ((float)dt / (float)(NUMDT - 1));
  float s = sinf(ang), c = cosf(ang);
  const float* it  = item_table + (size_t)item * DDIM;
  const float* dtt = dt_table   + (size_t)dt   * DDIM;
  __builtin_prefetch(it + threadIdx.x, 0, 1);          // global_prefetch_b8
  __builtin_prefetch(dtt + threadIdx.x, 0, 1);
  __bf16* out = token_bf + (size_t)row * DDIM;
  for (int d = threadIdx.x; d < DDIM; d += blockDim.x) {
    float t = it[d] + dtt[d] + s * Wc[2 * d] + c * Wc[2 * d + 1] + bc[d];
    out[d] = f2bf(t);
  }
}

// --------------------------- WMMA bf16 GEMM --------------------------------
// C[M,N] = A[M,512] * BT[N,512]^T (+bias[n]); one 16x16 tile per wave.
__global__ __launch_bounds__(256)
void gemm_bf16_wmma(const __bf16* __restrict__ A,
                    const __bf16* __restrict__ BT,
                    const float* __restrict__ bias,
                    float* __restrict__ C, int M, int N) {
  const int K = DDIM;
  int wave = (blockIdx.x * blockDim.x + threadIdx.x) >> 5;   // wave-uniform
  int lane = threadIdx.x & 31;
  int ntiles = N >> 4;
  int mtile = wave / ntiles, ntile = wave % ntiles;
  if (mtile >= (M >> 4)) return;                // uniform: EXEC stays all-1s
  int half = lane >> 4;
  int m = mtile * 16 + (lane & 15);
  int n = ntile * 16 + (lane & 15);
  v8f acc = {};
  for (int kk = 0; kk < K; kk += 32) {
    v16bf a, b;
    // A 16x32 bf16 layout: lane m holds K pairs; VGPR v: K = half*16 + v*2
#pragma unroll
    for (int v = 0; v < 8; ++v) {
      int bk = kk + (half << 4) + (v << 1);
      a[2 * v]     = A[(size_t)m * K + bk];
      a[2 * v + 1] = A[(size_t)m * K + bk + 1];
    }
    // B 32x16 bf16 layout: lane column n; 16 packed elems = K = half*16 + e
    const __bf16* bp = BT + (size_t)n * K + kk + (half << 4);
#pragma unroll
    for (int e = 0; e < 16; ++e) b[e] = bp[e];
    acc = __builtin_amdgcn_wmma_f32_16x16x32_bf16(
        false, a, false, b, (short)0, acc, false, false);
  }
  float bs = bias ? bias[n] : 0.0f;
#pragma unroll
  for (int r = 0; r < 8; ++r) {
    int row = mtile * 16 + half * 8 + r;        // C/D: M split at lane 16
    C[(size_t)row * N + n] = acc[r] + bs;
  }
}

// --------------------- attention: one workgroup per b ----------------------
// LDS: token[b] (200x512 bf16 = 200KB, fits WGP 320KB) + qk row + attn.
// Token tile fetched with the Tensor Data Mover (1-D D# descriptor).
__global__ __launch_bounds__(256)
void attention_kernel(const __bf16* __restrict__ token_bf,
                      const float* __restrict__ qk,
                      const int* __restrict__ mask,
                      const float* __restrict__ raw_tau,
                      float* __restrict__ attn_out,
                      __bf16* __restrict__ tbar_bf) {
  extern __shared__ char smem[];
  __bf16* tok_s  = (__bf16*)smem;                         // 200*512 bf16
  float*  qk_s   = (float*)(smem + LDIM * DDIM * 2);      // 512 f32
  float*  attn_s = qk_s + DDIM;                           // 256 f32 (200 used)
  int b = blockIdx.x, tid = threadIdx.x;

#if __has_builtin(__builtin_amdgcn_tensor_load_to_lds)
  if (tid < 32) {   // wave 0, full EXEC, uniform operands: one TDM issue
    const unsigned ELEMS = (LDIM * DDIM * 2) / 8;         // 25600 x 8B = 200KB
    unsigned long long ga =
        (unsigned long long)(size_t)(token_bf + (size_t)b * LDIM * DDIM);
    unsigned lds_addr = (unsigned)(size_t)(void*)tok_s;   // LDS offset = low 32b
    v4u g0; v8i g1; v4i g2 = {}; v4i g3 = {}; v8i g4 = {};
    // D# group0: count=1 | lds_addr | global_addr[56:0] | type=2
    g0[0] = 1u;
    g0[1] = lds_addr;
    g0[2] = (unsigned)(ga & 0xFFFFFFFFu);
    g0[3] = ((unsigned)((ga >> 32) & 0x01FFFFFFu)) | 0x80000000u;
    // D# group1: data_size=3 (8B), tensor_dim0=tile_dim0=25600 (1-D), dim1 unused
    g1[0] = 0x00030000;                                   // data_size=3
    g1[1] = (int)((ELEMS & 0xFFFFu) << 16);               // tensor_dim0[15:0]
    g1[2] = (int)(((ELEMS >> 16) & 0xFFFFu) | (1u << 16)); // dim0[31:16], tensor_dim1=1
    g1[3] = (int)((ELEMS & 0xFFFFu) << 16);               // tile_dim0
    g1[4] = 0;                                            // tile_dim1=0, tile_dim2=0
    g1[5] = (int)ELEMS;                                   // tensor_dim0_stride
    g1[6] = 0;
    g1[7] = 0;
    // 6-arg toolchain variant: groups 0..3, extra group (zero), cpol
    __builtin_amdgcn_tensor_load_to_lds(g0, g1, g2, g3, g4, 0);
    __builtin_amdgcn_s_wait_tensorcnt(0);
  }
  for (int d = tid; d < DDIM; d += 256) qk_s[d] = qk[b * DDIM + d];
#else
  const uint4* g = (const uint4*)(token_bf + (size_t)b * LDIM * DDIM);
  uint4* sm = (uint4*)tok_s;
  for (int i = tid; i < (LDIM * DDIM) / 8; i += 256) sm[i] = g[i];
  for (int d = tid; d < DDIM; d += 256) qk_s[d] = qk[b * DDIM + d];
#endif
  __syncthreads();

  float x = raw_tau[0];
  float tau = (x > 20.f ? x : log1pf(expf(x))) + 1e-6f;
  float inv_tau = 1.0f / tau;

  int wave = tid >> 5, lane = tid & 31;
  for (int l = wave * 25; l < wave * 25 + 25; ++l) {       // 8 waves * 25 = 200
    float p = 0.f;
#pragma unroll
    for (int j = 0; j < 16; ++j) {
      int d = lane + (j << 5);
      p += qk_s[d] * bf2f(tok_s[l * DDIM + d]);
    }
    for (int off = 16; off > 0; off >>= 1) p += __shfl_down(p, off, 32);
    if (lane == 0)
      attn_s[l] = mask[b * LDIM + l] ? p * inv_tau : -3.0e38f;
  }
  __syncthreads();

  if (wave == 0) {                                         // softmax over 200
    float mx = -3.4e38f;
    for (int l = lane; l < LDIM; l += 32) mx = fmaxf(mx, attn_s[l]);
    for (int off = 16; off > 0; off >>= 1) mx = fmaxf(mx, __shfl_xor(mx, off, 32));
    float ssum = 0.f;
    for (int l = lane; l < LDIM; l += 32) {
      float e = __expf(attn_s[l] - mx);
      attn_s[l] = e;
      ssum += e;
    }
    for (int off = 16; off > 0; off >>= 1) ssum += __shfl_xor(ssum, off, 32);
    float inv = 1.0f / ssum;
    for (int l = lane; l < LDIM; l += 32) {
      float a = attn_s[l] * inv;
      attn_s[l] = a;
      attn_out[b * LDIM + l] = a;
    }
  }
  __syncthreads();

  float a0 = 0.f, a1 = 0.f;                                // t_bar = attn . token
  for (int l = 0; l < LDIM; ++l) {
    float w = attn_s[l];
    a0 += w * bf2f(tok_s[l * DDIM + tid]);
    a1 += w * bf2f(tok_s[l * DDIM + tid + 256]);
  }
  tbar_bf[(size_t)b * DDIM + tid]       = f2bf(a0);
  tbar_bf[(size_t)b * DDIM + tid + 256] = f2bf(a1);
}

__global__ void tau_kernel(const float* __restrict__ raw_tau,
                           float* __restrict__ out) {
  float x = raw_tau[0];
  out[0] = (x > 20.f ? x : log1pf(expf(x))) + 1e-6f;
}

// ------------------------------- launcher ----------------------------------
extern "C" void kernel_launch(void* const* d_in, const int* in_sizes, int n_in,
                              void* d_out, int out_size, void* d_ws, size_t ws_size,
                              hipStream_t stream) {
  const int*   items      = (const int*)d_in[0];
  const int*   dts        = (const int*)d_in[1];
  const int*   mask       = (const int*)d_in[2];
  const int*   cand       = (const int*)d_in[3];
  const float* item_table = (const float*)d_in[4];
  const float* dt_table   = (const float*)d_in[5];
  const float* Wc         = (const float*)d_in[6];
  const float* bc         = (const float*)d_in[7];
  const float* Wk         = (const float*)d_in[8];
  // d_in[9] = bk: cancels in softmax, unused.
  const float* Wv         = (const float*)d_in[10];
  const float* bv         = (const float*)d_in[11];
  const float* raw_tau    = (const float*)d_in[12];

  float* u_out    = (float*)d_out;                         // [256,512]
  float* attn_out = u_out + BDIM * DDIM;                   // [256,200]
  float* tau_out  = attn_out + BDIM * LDIM;                // [1]

  char* ws = (char*)d_ws;
  size_t off = 0;
  __bf16* token_bf = (__bf16*)(ws + off); off += (size_t)BDIM * LDIM * DDIM * 2; // 52.4MB
  __bf16* WkT_bf   = (__bf16*)(ws + off); off += (size_t)DDIM * DDIM * 2;
  __bf16* Wv_bf    = (__bf16*)(ws + off); off += (size_t)DDIM * DDIM * 2;
  __bf16* q_bf     = (__bf16*)(ws + off); off += (size_t)BDIM * DDIM * 2;
  float*  qk       = (float*) (ws + off); off += (size_t)BDIM * DDIM * 4;
  __bf16* tbar_bf  = (__bf16*)(ws + off); off += (size_t)BDIM * DDIM * 2;

  prep_weights<<<1024, 256, 0, stream>>>(Wk, Wv, item_table, cand,
                                         WkT_bf, Wv_bf, q_bf);
  token_build<<<BDIM * LDIM, 256, 0, stream>>>(items, dts, item_table, dt_table,
                                               Wc, bc, token_bf);
  // qk = q @ Wk : 512 tiles -> 512 waves -> 64 blocks of 8 waves
  gemm_bf16_wmma<<<64, 256, 0, stream>>>(q_bf, WkT_bf, nullptr, qk, BDIM, DDIM);

  size_t smem = (size_t)LDIM * DDIM * 2 + (DDIM + 256) * 4; // ~207.9 KB < 320 KB
  attention_kernel<<<BDIM, 256, smem, stream>>>(token_bf, qk, mask, raw_tau,
                                                attn_out, tbar_bf);
  // u = t_bar @ Wv^T + bv
  gemm_bf16_wmma<<<64, 256, 0, stream>>>(tbar_bf, Wv_bf, bv, u_out, BDIM, DDIM);
  tau_kernel<<<1, 1, 0, stream>>>(raw_tau, tau_out);
}
